// PoolGCN_34033320853961
// MI455X (gfx1250) — compile-verified, matrix operand order checked by reading, not run on hardware
//
#include <hip/hip_runtime.h>
#include <math.h>

typedef __attribute__((ext_vector_type(16))) __bf16 v16bf;
typedef __attribute__((ext_vector_type(8)))  __bf16 v8bf;
typedef __attribute__((ext_vector_type(8)))  float  v8f;
typedef __attribute__((ext_vector_type(4)))  int    v4i;

#define BATCHN  8
#define TLEN    1024
#define NROWS   8192            // BATCHN*TLEN
#define MEMD    300
#define NHEADS  3
#define OT_ROWS 544             // >= max padded channel count (525 -> 544)
#define OT_BS   ((long)OT_ROWS*TLEN)
#define SLD     544             // leading dim of S (bf16, row-major, K-padded)

// Async global->LDS path (CDNA5). Guarded: falls back to load+ds_store if the
// toolchain doesn't expose the builtins (compile-safe either way).
#if defined(__AMDGCN__) && __has_builtin(__builtin_amdgcn_global_load_async_to_lds_b128) && \
    __has_builtin(__builtin_amdgcn_s_wait_asynccnt)
#define HAVE_ASYNC_LDS 1
typedef __attribute__((address_space(1))) v4i* gptr4;
typedef __attribute__((address_space(3))) v4i* lptr4;
#endif

// ---------------------------------------------------------------------------
// WMMA GEMM:  C(MxN) = A(MxK) @ Bt(NxK)^T  in bf16 with f32 accumulation.
// Block = 4 waves; each wave owns a 32x32 C tile (2x2 subtiles, 4 v_wmma per
// K-step); block tile = 128(M) x 32(N). The 32x32 bf16 B tile is staged into
// LDS once per K-step by the whole block (async-to-LDS when available),
// double-buffered and overlapped with compute. A streams straight from
// global (adjacency lives in the 192MB L2). K must be a multiple of 32.
// Epilogue: +AddT (transposed bf16 addend), +bias*bscale, *rowscale, relu;
// outputs: fp32 row-major, bf16 row-major, bf16 transposed (one 16B
// store/lane into the outT layout [batch][N][TLEN]).
// ---------------------------------------------------------------------------
__global__ __launch_bounds__(128)
void k_gemm_wmma(const __bf16* __restrict__ A, long aBS, int lda,
                 const __bf16* __restrict__ Bt, long bBS, int ldb,
                 int Mper, int N, int K,
                 float* __restrict__ Cf, int ldc,
                 __bf16* __restrict__ Cb, int ldcb,
                 __bf16* __restrict__ Ct, int ctOff,
                 const __bf16* __restrict__ AddT,
                 const float* __restrict__ bias, float bscale,
                 const float* __restrict__ rs, int relu)
{
    __shared__ __attribute__((aligned(16))) __bf16 sB[2][32*32];

    const int tid  = threadIdx.x;        // 0..127
    const int wid  = tid >> 5;           // wave 0..3
    const int lane = tid & 31;
    const int hl   = lane >> 4;          // K-half selector
    const int l16  = lane & 15;
    const int n0   = blockIdx.x << 5;    // 32-wide N tile (shared by block)
    const int m0w  = (blockIdx.y << 7) + (wid << 5);  // wave's 32-row M tile
    const int bz   = blockIdx.z;

    // A fragment pointers for the wave's two 16-row subtiles
    const __bf16* Ap0 = A + (long)bz*aBS + (long)(m0w + l16)*lda + hl*8;
    const __bf16* Ap1 = Ap0 + (long)16*lda;

    // B staging: each of 128 threads moves one 16B chunk per K-step
    const int srow  = tid >> 2;          // 0..31  (row of Bt tile)
    const int skoff = (tid & 3) * 8;     // 0,8,16,24 (bf16 elems within chunk)
    const __bf16* gB = Bt + (long)bz*bBS + (long)(n0 + srow)*ldb + skoff;

    auto stage = [&](int buf, int k) {
#ifdef HAVE_ASYNC_LDS
        __builtin_amdgcn_global_load_async_to_lds_b128(
            (gptr4)(gB + k),
            (lptr4)&sB[buf][srow*32 + skoff],
            0, 0);
#else
        *(v8bf*)&sB[buf][srow*32 + skoff] = *(const v8bf*)(gB + k);
#endif
    };

    v8f acc[2][2] = {};
    stage(0, 0);
#ifdef HAVE_ASYNC_LDS
    __builtin_amdgcn_s_wait_asynccnt(0);
#endif
    __syncthreads();

    int cur = 0;
    for (int k = 0; k < K; k += 32) {
        const int nxt = cur ^ 1;
        if (k + 32 < K) stage(nxt, k + 32);

        union U { v16bf v; v8bf h[2]; } a0, a1, b0, b1;
        a0.h[0] = *(const v8bf*)(Ap0 + k);
        a0.h[1] = *(const v8bf*)(Ap0 + k + 16);
        a1.h[0] = *(const v8bf*)(Ap1 + k);
        a1.h[1] = *(const v8bf*)(Ap1 + k + 16);
        __builtin_prefetch(Ap0 + k + 256, 0, 1);
        __builtin_prefetch(Ap1 + k + 256, 0, 1);
        const __bf16* sc = sB[cur];
        b0.h[0] = *(const v8bf*)(sc + l16*32 + hl*8);
        b0.h[1] = *(const v8bf*)(sc + l16*32 + hl*8 + 16);
        b1.h[0] = *(const v8bf*)(sc + (16 + l16)*32 + hl*8);
        b1.h[1] = *(const v8bf*)(sc + (16 + l16)*32 + hl*8 + 16);

        acc[0][0] = __builtin_amdgcn_wmma_f32_16x16x32_bf16(false, a0.v, false, b0.v,
                                                            (short)0, acc[0][0], false, false);
        acc[0][1] = __builtin_amdgcn_wmma_f32_16x16x32_bf16(false, a0.v, false, b1.v,
                                                            (short)0, acc[0][1], false, false);
        acc[1][0] = __builtin_amdgcn_wmma_f32_16x16x32_bf16(false, a1.v, false, b0.v,
                                                            (short)0, acc[1][0], false, false);
        acc[1][1] = __builtin_amdgcn_wmma_f32_16x16x32_bf16(false, a1.v, false, b1.v,
                                                            (short)0, acc[1][1], false, false);
#ifdef HAVE_ASYNC_LDS
        __builtin_amdgcn_s_wait_asynccnt(0);
#endif
        __syncthreads();
        cur = nxt;
    }

    // ------------------------------ epilogue ------------------------------
    #pragma unroll
    for (int i = 0; i < 2; ++i) {
        const int  mb   = m0w + 16*i + hl*8;        // first of 8 contiguous rows
        const long grow = (long)bz*Mper + mb;       // global row index
        const int  bt   = (int)(grow >> 10);        // 1024-row batch index
        const int  ml   = (int)(grow & 1023);
        #pragma unroll
        for (int j = 0; j < 2; ++j) {
            const int n = n0 + 16*j + l16;
            v8f a = acc[i][j];
            if (AddT) {
                v8bf ad = *(const v8bf*)(AddT + (long)bt*OT_BS + (long)n*TLEN + ml);
                #pragma unroll
                for (int r = 0; r < 8; ++r) a[r] += (float)ad[r];
            }
            const float bvv = (bias && n < N) ? bscale * bias[n] : 0.f;
            float o[8];
            #pragma unroll
            for (int r = 0; r < 8; ++r) {
                float v = a[r] + bvv;
                if (rs) v *= rs[grow + r];
                if (relu && v < 0.f) v = 0.f;
                o[r] = v;
            }
            if (n < N) {
                if (Cf) {
                    #pragma unroll
                    for (int r = 0; r < 8; ++r) Cf[(grow + r)*(long)ldc + n] = o[r];
                }
                if (Cb) {
                    #pragma unroll
                    for (int r = 0; r < 8; ++r) Cb[(grow + r)*(long)ldcb + n] = (__bf16)o[r];
                }
                if (Ct) {
                    v8bf t;
                    #pragma unroll
                    for (int r = 0; r < 8; ++r) t[r] = (__bf16)o[r];
                    *(v8bf*)(Ct + (long)bt*OT_BS + (long)(ctOff + n)*TLEN + ml) = t;
                }
            }
        }
    }
}

// --------------------------- elementwise helpers ---------------------------

// fp32 [M][cols] -> bf16 [M][Kpad], zero pad cols
__global__ void k_pack1(__bf16* dst, const float* src, int M, int cols, int Kpad) {
    long g = (long)blockIdx.x*blockDim.x + threadIdx.x;
    if (g >= (long)M*Kpad) return;
    int k = (int)(g % Kpad); long m = g / Kpad;
    dst[g] = (k < cols) ? (__bf16)src[m*cols + k] : (__bf16)0.f;
}

// concat [pool(300) | x(300)] -> bf16 [NROWS][608]
__global__ void k_pack2cat(__bf16* dst, const float* a, const float* b) {
    long g = (long)blockIdx.x*blockDim.x + threadIdx.x;
    if (g >= (long)NROWS*608) return;
    int k = (int)(g % 608); long m = g / 608;
    float v = 0.f;
    if (k < 300)      v = a[m*300 + k];
    else if (k < 600) v = b[m*300 + k - 300];
    dst[g] = (__bf16)v;
}

// bf16(headcat(900) + tile3(xin,300)) -> [NROWS][928]
__global__ void k_pack_hc(__bf16* dst, const float* hc, const float* xin) {
    long g = (long)blockIdx.x*blockDim.x + threadIdx.x;
    if (g >= (long)NROWS*928) return;
    int k = (int)(g % 928); long m = g / 928;
    float v = 0.f;
    if (k < 900) v = hc[m*900 + k] + xin[m*300 + (k % 300)];
    dst[g] = (__bf16)v;
}

// concat [layer1(300) | rep3(tmp2,100)(300)] -> bf16 [NROWS][608]
__global__ void k_pack_agg(__bf16* dst, const float* lay1, const float* t2) {
    long g = (long)blockIdx.x*blockDim.x + threadIdx.x;
    if (g >= (long)NROWS*608) return;
    int k = (int)(g % 608); long m = g / 608;
    float v = 0.f;
    if (k < 300)      v = lay1[m*300 + k];
    else if (k < 600) v = t2[m*100 + (k - 300) % 100];
    dst[g] = (__bf16)v;
}

// fp32 W [K][N] -> bf16 Wt [N][Kpad] (B operand), zero pad cols
__global__ void k_packT_w(__bf16* dst, const float* src, int K, int N, int Kpad) {
    long g = (long)blockIdx.x*blockDim.x + threadIdx.x;
    if (g >= (long)N*Kpad) return;
    int k = (int)(g % Kpad); long n = g / Kpad;
    dst[g] = (k < K) ? (__bf16)src[(long)k*N + n] : (__bf16)0.f;
}

// fp32 act [NROWS][300] -> transposed bf16 into outT rows [0..300)
__global__ void k_packT300(__bf16* ot, const float* src) {
    long g = (long)blockIdx.x*blockDim.x + threadIdx.x;
    if (g >= (long)NROWS*300) return;
    int c = (int)(g % 300); long m = g / 300;
    ot[(long)(m >> 10)*OT_BS + (long)c*TLEN + (m & (TLEN-1))] = (__bf16)src[g];
}

// layer1[m][c] = tmp[m][c%100]
__global__ void k_rep3(float* dst, const float* src) {
    long g = (long)blockIdx.x*blockDim.x + threadIdx.x;
    if (g >= (long)NROWS*300) return;
    dst[g] = src[(g/300)*100 + (g % 300) % 100];
}

__global__ void k_zero_cols(__bf16* p, int ld, int rows, int c0, int w) {
    long g = (long)blockIdx.x*blockDim.x + threadIdx.x;
    if (g >= (long)rows*w) return;
    long r = g / w; int c = c0 + (int)(g % w);
    p[r*ld + c] = (__bf16)0.f;
}

// max-pool window [t-4, t+4] clamped within each 1024-row batch
__global__ void k_pool(float* xp, const float* x) {
    long g = (long)blockIdx.x*blockDim.x + threadIdx.x;
    if (g >= (long)NROWS*MEMD) return;
    int c = (int)(g % MEMD); long m = g / MEMD;
    int t = (int)(m & (TLEN-1));
    long base = m - t;
    int lo = t - 4 < 0 ? 0 : t - 4;
    int hi = t + 4 > TLEN-1 ? TLEN-1 : t + 4;
    float v = -3.4e38f;
    for (int tt = lo; tt <= hi; ++tt) {
        float u = x[(base + tt)*MEMD + c];
        v = u > v ? u : v;
    }
    xp[g] = v;
}

// gp = x_new @ W(300x6) + b   (tiny N -> plain VALU dot)
__global__ void k_gp(float* gp, const float* xn, const float* W, const float* b) {
    int g = blockIdx.x*blockDim.x + threadIdx.x;
    if (g >= NROWS*6) return;
    int m = g / 6, j = g % 6;
    const float* xr = xn + (long)m*MEMD;
    float s = b[j];
    for (int k = 0; k < MEMD; ++k) s += xr[k]*W[k*6 + j];
    gp[g] = s;
}

// per-node stats: mean, std^2 = softplus(.)^2, log(std)
__global__ void k_stats(float* mean, float* s2, float* ls, const float* gp) {
    int m = blockIdx.x*blockDim.x + threadIdx.x;
    if (m >= NROWS) return;
    for (int h = 0; h < NHEADS; ++h) {
        float mu = gp[m*6 + h];
        float x  = gp[m*6 + 3 + h];
        float sp = (x > 20.f) ? x : log1pf(expf(x));
        mean[h*NROWS + m] = mu;
        s2[h*NROWS + m]   = sp*sp;
        ls[h*NROWS + m]   = logf(sp);
    }
}

// adj[h][b][i][j] = 0.5*(2 ls_i - 2 ls_j + (s2_j + (m_j-m_i)^2)/s2_i - 1), bf16
__global__ void k_adjgen(__bf16* adj, const float* mean, const float* s2, const float* ls) {
    long g = (long)blockIdx.x*blockDim.x + threadIdx.x;
    long total = (long)NHEADS*NROWS*(TLEN/4);
    if (g >= total) return;
    int  j0     = (int)(g % (TLEN/4)) * 4;
    long row    = g / (TLEN/4);                 // [0, 3*8192)
    int  node_i = (int)(row % NROWS);
    int  h      = (int)(row / NROWS);
    int  jbase  = h*NROWS + (node_i & ~(TLEN-1));
    float mi  = mean[h*NROWS + node_i];
    float inv = 1.0f / s2[h*NROWS + node_i];
    float lsi = ls[h*NROWS + node_i];
    __bf16* out = adj + row*TLEN + j0;
    for (int jj = 0; jj < 4; ++jj) {
        int jn = jbase + j0 + jj;
        float dm = mean[jn] - mi;
        float v  = 0.5f*(2.f*lsi - 2.f*ls[jn] + (s2[jn] + dm*dm)*inv - 1.f);
        out[jj] = (__bf16)v;
    }
}

// rowscale[h*8192+m] = 1/(sum_j adj + 1): one wave per row, shfl_xor reduce
__global__ __launch_bounds__(256)
void k_rowscale(float* rs, const __bf16* adj, int nrows) {
    int row  = blockIdx.x*8 + (threadIdx.x >> 5);
    int lane = threadIdx.x & 31;
    if (row >= nrows) return;
    const __bf16* p = adj + (long)row*TLEN;
    float s = 0.f;
    for (int c = lane; c < TLEN; c += 32) s += (float)p[c];
    for (int off = 16; off > 0; off >>= 1) s += __shfl_xor(s, off, 32);
    if (lane == 0) rs[row] = 1.0f / (s + 1.0f);
}

// ---------------------------------------------------------------------------

extern "C" void kernel_launch(void* const* d_in, const int* in_sizes, int n_in,
                              void* d_out, int out_size, void* d_ws, size_t ws_size,
                              hipStream_t stream)
{
    (void)out_size; (void)ws_size; (void)n_in;

    // ---- resolve flatten order of the nested params dict ----
    bool sorted = (in_sizes[1] == NROWS);   // jax-style sorted keys: input_id is 2nd
    int I_inputs, I_w_in, I_ggg_t, I_ggg_g, I_mgc1, I_mgc1_out,
        I_sag1_lt, I_mgc2, I_mgc2_out, I_sag2_lt, I_agg;
    if (!sorted) {   // insertion order
        I_inputs=1;  I_w_in=3;  I_ggg_t=5;  I_ggg_g=7;  I_mgc1=9;  I_mgc1_out=21;
        I_sag1_lt=27; I_mgc2=29; I_mgc2_out=53; I_sag2_lt=59; I_agg=61;
    } else {         // alphabetically sorted keys
        I_inputs=2;  I_agg=3;  I_ggg_g=5;  I_ggg_t=7;  I_mgc1=9;  I_mgc1_out=21;
        I_mgc2=23; I_mgc2_out=47; I_sag1_lt=49; I_sag2_lt=55; I_w_in=61;
    }
    auto F = [&](int i) { return (const float*)d_in[i]; };

    // ---- carve workspace ----
    char* base = (char*)d_ws;
    size_t off = 0;
    auto carve = [&](size_t bytes) {
        void* p = base + off;
        off = (off + bytes + 255) & ~(size_t)255;
        return p;
    };
    __bf16* adjB = (__bf16*)carve((size_t)NHEADS*BATCHN*TLEN*TLEN*2); // 50.3 MB
    __bf16* outT = (__bf16*)carve((size_t)BATCHN*OT_ROWS*TLEN*2);     // 8.9 MB
    __bf16* Abuf = (__bf16*)carve((size_t)NROWS*928*2);               // 15.2 MB
    __bf16* Sb   = (__bf16*)carve((size_t)NROWS*SLD*2);               // 8.9 MB
    __bf16* WtB  = (__bf16*)carve((size_t)320*928*2);                 // 32-row N tiles
    float*  xF   = (float*)carve((size_t)NROWS*MEMD*4);
    float*  xpF  = (float*)carve((size_t)NROWS*MEMD*4);
    float*  xnF  = (float*)carve((size_t)NROWS*MEMD*4);
    float*  gpF  = (float*)carve((size_t)NROWS*6*4);
    float*  mA   = (float*)carve((size_t)NHEADS*NROWS*4);
    float*  s2A  = (float*)carve((size_t)NHEADS*NROWS*4);
    float*  lsA  = (float*)carve((size_t)NHEADS*NROWS*4);
    float*  rsA  = (float*)carve((size_t)NHEADS*NROWS*4);
    float*  hcF  = (float*)carve((size_t)NROWS*900*4);
    float*  out1F= (float*)carve((size_t)NROWS*MEMD*4);
    float*  tmp1F= (float*)carve((size_t)NROWS*100*4);
    float*  lay1F= (float*)carve((size_t)NROWS*MEMD*4);
    float*  out2F= (float*)carve((size_t)NROWS*MEMD*4);
    float*  tmp2F= (float*)carve((size_t)NROWS*100*4);

    auto up32 = [](int v) { return (v + 31) & ~31; };
    auto g1   = [](long n) { return dim3((unsigned)((n + 255) / 256)); };

    auto gemm = [&](const __bf16* A, long aBS, int lda,
                    const __bf16* Bt, long bBS, int ldb,
                    int Mper, int N, int K, int batches,
                    float* Cf, int ldc, __bf16* Cb, int ldcb,
                    __bf16* Ct, int ctOff, const __bf16* AddT,
                    const float* bias, float bscale, const float* rs, int relu) {
        dim3 grid((N + 31) / 32, Mper / 128, batches);
        k_gemm_wmma<<<grid, 128, 0, stream>>>(A, aBS, lda, Bt, bBS, ldb, Mper, N, K,
                                              Cf, ldc, Cb, ldcb, Ct, ctOff, AddT,
                                              bias, bscale, rs, relu);
    };

    // ================= stage 0: input projection =================
    k_pack1<<<g1((long)NROWS*768), 256, 0, stream>>>(Abuf, F(I_inputs), NROWS, 768, 768);
    k_packT_w<<<g1((long)300*768), 256, 0, stream>>>(WtB, F(I_w_in), 768, 300, 768);
    gemm(Abuf, 0, 768, WtB, 0, 768, NROWS, 300, 768, 1,
         xF, 300, nullptr, 0, nullptr, 0, nullptr, F(I_w_in+1), 1.f, nullptr, 0);

    // ================= _ggg =================
    k_pool<<<g1((long)NROWS*MEMD), 256, 0, stream>>>(xpF, xF);
    k_pack2cat<<<g1((long)NROWS*608), 256, 0, stream>>>(Abuf, xpF, xF);
    k_packT_w<<<g1((long)300*608), 256, 0, stream>>>(WtB, F(I_ggg_t), 600, 300, 608);
    // x_new: fp32 out + transposed bf16 into outT rows [0..300) (mgc1's x_in)
    gemm(Abuf, 0, 608, WtB, 0, 608, NROWS, 300, 608, 1,
         xnF, 300, nullptr, 0, outT, 0, nullptr, F(I_ggg_t+1), 1.f, nullptr, 0);

    k_gp<<<g1((long)NROWS*6), 256, 0, stream>>>(gpF, xnF, F(I_ggg_g), F(I_ggg_g+1));
    k_stats<<<g1((long)NROWS), 256, 0, stream>>>(mA, s2A, lsA, gpF);
    k_adjgen<<<g1((long)NHEADS*NROWS*(TLEN/4)), 256, 0, stream>>>(adjB, mA, s2A, lsA);
    k_rowscale<<<(NHEADS*NROWS)/8, 256, 0, stream>>>(rsA, adjB, NHEADS*NROWS);

    // ================= two MGC stages =================
    for (int stage = 0; stage < 2; ++stage) {
        const int   L    = (stage == 0) ? 2 : 4;
        const int   HD   = (stage == 0) ? 150 : 75;
        const float* xin = (stage == 0) ? xnF : lay1F;
        const int wbase  = (stage == 0) ? I_mgc1 : I_mgc2;
        const int wout   = (stage == 0) ? I_mgc1_out : I_mgc2_out;
        const int wlt    = (stage == 0) ? I_sag1_lt : I_sag2_lt;
        float* outF      = (stage == 0) ? out1F : out2F;
        float* tmpF      = (stage == 0) ? tmp1F : tmp2F;

        for (int h = 0; h < NHEADS; ++h) {
            int C = MEMD;
            for (int l = 0; l < L; ++l) {
                const int Cp = up32(C);
                if (Cp > C)   // zero K-pad cols of S for the following GEMM
                    k_zero_cols<<<g1((long)NROWS*(Cp - C)), 256, 0, stream>>>(
                        Sb, SLD, NROWS, C, Cp - C);

                // S = adj[h] @ outputs + outputs  (per-batch 1024x1024 @ 1024xC)
                gemm(adjB + (long)h*BATCHN*TLEN*TLEN, (long)TLEN*TLEN, TLEN,
                     outT, OT_BS, TLEN, TLEN, C, TLEN, BATCHN,
                     nullptr, 0, Sb, SLD, nullptr, 0, outT,
                     nullptr, 0.f, nullptr, 0);

                // g = relu((S @ W + 2b) / denom); write fp32 into headcat slice
                // and transposed bf16 into outT rows [C..C+HD) for the next layer
                const float* W  = F(wbase + 2*(h*L + l));
                const float* bb = F(wbase + 2*(h*L + l) + 1);
                k_packT_w<<<g1((long)HD*Cp), 256, 0, stream>>>(WtB, W, C, HD, Cp);
                gemm(Sb, 0, SLD, WtB, 0, Cp, NROWS, HD, Cp, 1,
                     hcF + h*MEMD + l*HD, 900, nullptr, 0, outT, C, nullptr,
                     bb, 2.f, rsA + h*NROWS, 1);
                C += HD;
            }
        }

        // head_out concat (+x_in) -> mgc_out GEMM (K=900)
        k_pack_hc<<<g1((long)NROWS*928), 256, 0, stream>>>(Abuf, hcF, xin);
        k_packT_w<<<g1((long)300*928), 256, 0, stream>>>(WtB, F(wout), 900, 300, 928);
        gemm(Abuf, 0, 928, WtB, 0, 928, NROWS, 300, 928, 1,
             outF, 300, nullptr, 0, nullptr, 0, nullptr, F(wout+1), 1.f, nullptr, 0);

        // sagpool live path: tmp = out @ lt + b  (300 -> 100)
        k_pack1<<<g1((long)NROWS*320), 256, 0, stream>>>(Abuf, outF, NROWS, 300, 320);
        k_packT_w<<<g1((long)100*320), 256, 0, stream>>>(WtB, F(wlt), 300, 100, 320);
        gemm(Abuf, 0, 320, WtB, 0, 320, NROWS, 100, 320, 1,
             tmpF, 100, nullptr, 0, nullptr, 0, nullptr, F(wlt+1), 1.f, nullptr, 0);

        if (stage == 0) {
            // layer1 = rep3(tmp1); becomes mgc2's x_in: fp32 + outT rows [0..300)
            k_rep3<<<g1((long)NROWS*300), 256, 0, stream>>>(lay1F, tmp1F);
            k_packT300<<<g1((long)NROWS*300), 256, 0, stream>>>(outT, lay1F);
        }
    }

    // ================= final agg GEMM (K=600) -> d_out =================
    k_pack_agg<<<g1((long)NROWS*608), 256, 0, stream>>>(Abuf, lay1F, tmp2F);
    k_packT_w<<<g1((long)300*608), 256, 0, stream>>>(WtB, F(I_agg), 600, 300, 608);
    gemm(Abuf, 0, 608, WtB, 0, 608, NROWS, 300, 608, 1,
         (float*)d_out, 300, nullptr, 0, nullptr, 0, nullptr,
         F(I_agg+1), 1.f, nullptr, 0);
}
